// Mamba2Layer_88304527606391
// MI455X (gfx1250) — matrix operand order repaired and checked by reading, not compiled, not run on hardware
//
#include <hip/hip_runtime.h>
#include <hip/hip_bf16.h>

// ---------------------------------------------------------------------------
// MI455X (gfx1250): all large matmuls via v_wmma_f32_16x16x32_bf16 (wave32,
// f32 accumulate). Bandwidth-bound workload (~224 GFLOP vs ~0.5GB @23.3TB/s):
//  - bf16 operands, b128 vectorized fragment loads (bit-cast to v16bf)
//  - weights pre-packed into fragment-native layout (coalesced b128 loads)
//  - 16x(16*NC) strip per wave, NC compile-time: straight-line WMMA chain,
//    one A fragment feeds NC WMMAs, loads clustered ahead of the WMMAs
//  - fused epilogues: silu/sigmoid-gate/bias/mask/mem-gate/transposed & packed
// ---------------------------------------------------------------------------

#define BATCHN 8
#define SEQ 1024
#define DMODEL 1024
#define NST 16
#define BSROWS (BATCHN * SEQ) // 8192

typedef __bf16 bf16;
typedef __attribute__((ext_vector_type(16))) __bf16 v16bf;
typedef __attribute__((ext_vector_type(8))) float v8f;
typedef __attribute__((ext_vector_type(4))) unsigned int v4u;
typedef __attribute__((ext_vector_type(8))) unsigned int v8u;

__device__ __forceinline__ float sigmoidf_(float x) { return 1.f / (1.f + __expf(-x)); }

// two contiguous 16-byte runs -> one v16bf fragment
__device__ __forceinline__ v16bf load_frag(const bf16* p0, const bf16* p1)
{
    v4u lo = *(const v4u*)p0;
    v4u hi = *(const v4u*)p1;
    v8u u;
    u[0] = lo[0]; u[1] = lo[1]; u[2] = lo[2]; u[3] = lo[3];
    u[4] = hi[0]; u[5] = hi[1]; u[6] = hi[2]; u[7] = hi[3];
    return __builtin_bit_cast(v16bf, u);
}

enum Ep { EP_F32, EP_SILU_BF16, EP_BF16, EP_GATE_ACC, EP_MASK, EP_MEMGATE, EP_PACKV };

// ---------------------------------------------------------------------------
// WMMA GEMM. A[M,K] bf16 row-major. B operand:
//   BM==0 : packed fragment-native layout, per 32k x 16n block:
//           off = ((nTile*(K/32)+kBlk)<<9) + (lane<<4) + j   (lane = n + 16*(k>>4), j = k&15, k/n block-local)
//   BM==1 : B is [N,K] row-major (B^T access, Q·K^T) -> contiguous runs
// Each wave computes a 16 x (16*NC) strip; NC is compile-time so the WMMA
// chain is branch-free (EXEC stays all-1s, accumulators stay pinned).
// ---------------------------------------------------------------------------
template <Ep EP, int BM, bool STORE_T, int NC>
__global__ void __launch_bounds__(128) gemm_wmma(
    const bf16* __restrict__ A, const bf16* __restrict__ B,
    float* __restrict__ outF, bf16* __restrict__ outB,
    const float* __restrict__ bias,
    const float* __restrict__ aux0, const float* __restrict__ aux1,
    const unsigned char* __restrict__ mask,
    int M, int N, int K, int lda, int ldb, int ldc,
    long sA, long sB, long sC, long sMask,
    float scale, int accumulate)
{
    const int lane = threadIdx.x & 31;
    const int wave = threadIdx.x >> 5;
    const int tilesN = N >> 4;
    const int stripsN = tilesN / NC;            // exact by construction
    const int numStrips = (M >> 4) * stripsN;
    const int strip = blockIdx.x * 4 + wave;
    if (strip >= numStrips) return;             // whole-wave exit: EXEC stays all-1s
    const int row0 = (strip / stripsN) << 4;
    const int colT0 = (strip % stripsN) * NC;   // first N-tile of the strip
    const long bz = blockIdx.z;

    const bf16* Ab = A + bz * sA;
    const bf16* Bb = B + bz * sB;

    const int lm = lane & 15;   // A row-in-tile / B col-in-tile / C col
    const int lh = lane >> 4;   // lane half

    // A layout: fragment = [row, k0 + lh*16 .. +16)
    const bf16* ap = Ab + (long)(row0 + lm) * lda + (lh << 4);
    const int kblocks = K >> 5;

    // per-column B base pointers (k0-invariant part hoisted)
    const bf16* bbase[NC];
#pragma unroll
    for (int c = 0; c < NC; ++c) {
        if (BM == 0)
            bbase[c] = Bb + (((long)(colT0 + c) * kblocks) << 9) + (lane << 4);
        else
            bbase[c] = Bb + (long)(((colT0 + c) << 4) + lm) * ldb + (lh << 4);
    }

    v8f acc[NC] = {};
    for (int k0 = 0; k0 < K; k0 += 32) {
        const v16bf af = load_frag(ap + k0, ap + k0 + 8);
        v16bf bfr[NC];
#pragma unroll
        for (int c = 0; c < NC; ++c) {
            const bf16* q = (BM == 0) ? (bbase[c] + ((long)(k0 >> 5) << 9))
                                      : (bbase[c] + k0);
            bfr[c] = load_frag(q, q + 8);
        }
#pragma unroll
        for (int c = 0; c < NC; ++c)
            acc[c] = __builtin_amdgcn_wmma_f32_16x16x32_bf16(
                false, af, false, bfr[c], (short)0, acc[c], false, false);
        if (k0 + 32 < K)
            __builtin_prefetch((const void*)(ap + k0 + 32), 0, 0); // global_prefetch_b8
    }

    // C layout: lane L -> N = L&15 ; VGPR v -> M = v + 8*(L>>4)
    const int rbase = row0 + (lh << 3);
#pragma unroll
    for (int c = 0; c < NC; ++c) {
        const int cc = ((colT0 + c) << 4) + lm;
        const float bval = bias ? bias[cc] : 0.f;
#pragma unroll
        for (int v = 0; v < 8; ++v) {
            const int rr = rbase + v;
            float a = acc[c][v] + bval;
            const long idx = STORE_T ? (bz * sC + (long)cc * ldc + rr)
                                     : (bz * sC + (long)rr * ldc + cc);
            if (EP == EP_F32) {
                outF[idx] = a;
            } else if (EP == EP_BF16) {
                outB[idx] = (bf16)a;
            } else if (EP == EP_SILU_BF16) {
                outB[idx] = (bf16)(a * sigmoidf_(a));
            } else if (EP == EP_GATE_ACC) {
                float prev = accumulate ? outF[idx] : 0.f;
                outF[idx] = prev + aux0[idx] * sigmoidf_(a) * aux1[0];
            } else if (EP == EP_MASK) {
                unsigned char mk = mask[bz * sMask + (long)rr * ldc + cc];
                outF[idx] = mk ? (a * scale) : -1e30f;
            } else if (EP == EP_MEMGATE) {
                outF[idx] = a * aux0[idx] * aux1[cc]; // (acc+bo) * comb * mem_gate
            } else if (EP == EP_PACKV) {
                // scatter into packed-B layout (per batch, K=SEQ rows)
                const int b = rr >> 10, kk = rr & (SEQ - 1);
                const long off = (long)b * SEQ * DMODEL
                               + (((long)(cc >> 4) * (SEQ >> 5) + (kk >> 5)) << 9)
                               + (((cc & 15) + (((kk >> 4) & 1) << 4)) << 4) + (kk & 15);
                outB[off] = (bf16)a;
            }
        }
    }
}

// ---------------------------------------------------------------------------
__global__ void f32_to_bf16_k(const float* __restrict__ in, bf16* __restrict__ out, long n)
{
    long i = (long)blockIdx.x * blockDim.x + threadIdx.x;
    if (i < n) out[i] = (bf16)in[i];
}

// pack a [K,N] row-major f32 matrix into the fragment-native bf16 B layout
__global__ void packB_k(const float* __restrict__ in, bf16* __restrict__ out, int K, int N)
{
    long i = (long)blockIdx.x * blockDim.x + threadIdx.x;
    if (i >= (long)K * N) return;
    const int k = (int)(i / N), n = (int)(i % N);
    const long off = (((long)(n >> 4) * (K >> 5) + (k >> 5)) << 9)
                   + (((n & 15) + (((k >> 4) & 1) << 4)) << 4) + (k & 15);
    out[off] = (bf16)in[i];
}

// mean over seq axis: [B,S,D] -> [B,D]
__global__ void mean_seq_k(const float* __restrict__ in, float* __restrict__ out)
{
    int i = blockIdx.x * blockDim.x + threadIdx.x;
    if (i >= BATCHN * DMODEL) return;
    int b = i / DMODEL, d = i % DMODEL;
    const float* p = in + (long)b * SEQ * DMODEL + d;
    float s = 0.f;
    for (int t = 0; t < SEQ; ++t) s += p[(long)t * DMODEL];
    out[i] = s * (1.f / SEQ);
}

// cf = sigmoid( silu(mean @ w1 + b1) @ w2 + b2 )   grid=(B, T), block=H
__global__ void gate_mlp_k(const float* __restrict__ mean,
                           const float* __restrict__ w1, const float* __restrict__ b1,
                           const float* __restrict__ w2, const float* __restrict__ b2,
                           float* __restrict__ cf, int H)
{
    extern __shared__ float sm[];
    const int b = blockIdx.x, t = blockIdx.y, j = threadIdx.x;
    const float* mp = mean + b * DMODEL;
    const float* w1p = w1 + (long)t * DMODEL * H;
    float acc = b1[t * H + j];
    for (int d = 0; d < DMODEL; ++d) acc += mp[d] * w1p[(long)d * H + j];
    float h = acc * sigmoidf_(acc); // silu
    sm[j] = h * w2[t * H + j];
    __syncthreads();
    for (int st = H >> 1; st > 0; st >>= 1) {
        if (j < st) sm[j] += sm[j + st];
        __syncthreads();
    }
    if (j == 0) cf[t * gridDim.x + b] = sigmoidf_(sm[0] + b2[t]);
}

// sequential SSM scan over seq: h = tanh(h@A) + tanh(cf*U);  grid=24, block=32
__global__ void ssm_scan_k(const float* __restrict__ U, const float* __restrict__ Amat,
                           const float* __restrict__ cf, float* __restrict__ H)
{
    const int tb = blockIdx.x;          // t*8 + b
    const int t = tb >> 3;
    const int lane = threadIdx.x;
    __shared__ float h[NST];
    __shared__ float As[NST * NST];
    for (int i = lane; i < NST * NST; i += 32) As[i] = Amat[t * NST * NST + i];
    if (lane < NST) h[lane] = 0.f;
    const float c = cf[tb];
    const float* Up = U + (long)tb * SEQ * NST;
    float* Hp = H + (long)tb * SEQ * NST;
    __syncthreads();
    for (int s = 0; s < SEQ; ++s) {
        float hn = 0.f;
        if (lane < NST) {
            float acc = 0.f;
#pragma unroll
            for (int m = 0; m < NST; ++m) acc += h[m] * As[m * NST + lane];
            hn = tanhf(acc) + tanhf(c * Up[s * NST + lane]);
        }
        __syncthreads();
        if (lane < NST) { h[lane] = hn; Hp[s * NST + lane] = hn; }
        __syncthreads();
    }
}

// comb = sum_t tw[t] * ( tanh(H@C) + Dv*cf*gated ), tw = softmax(ts_logits)
__global__ void ssm_combine_k(const float* __restrict__ H, const float* __restrict__ Cmat,
                              const float* __restrict__ Dv, const float* __restrict__ cf,
                              const float* __restrict__ gated, const float* __restrict__ tsl,
                              float* __restrict__ comb, bf16* __restrict__ comb_bf)
{
    long i = (long)blockIdx.x * blockDim.x + threadIdx.x;
    if (i >= (long)BSROWS * DMODEL) return;
    const int d = (int)(i & (DMODEL - 1));
    const long bs = i >> 10;
    const int s = (int)(bs & (SEQ - 1));
    const int b = (int)(bs >> 10);
    float l0 = tsl[0], l1 = tsl[1], l2 = tsl[2];
    float mx = fmaxf(l0, fmaxf(l1, l2));
    float e0 = __expf(l0 - mx), e1 = __expf(l1 - mx), e2 = __expf(l2 - mx);
    float inv = 1.f / (e0 + e1 + e2);
    float tw[3] = { e0 * inv, e1 * inv, e2 * inv };
    float g = gated[i];
    float r = 0.f;
#pragma unroll
    for (int t = 0; t < 3; ++t) {
        const float* Hp = H + ((long)(t * 8 + b) * SEQ + s) * NST;
        const float* Cp = Cmat + (long)t * NST * DMODEL + d;
        float acc = 0.f;
#pragma unroll
        for (int n = 0; n < NST; ++n) acc += Hp[n] * Cp[(long)n * DMODEL];
        r += tw[t] * (tanhf(acc) + Dv[t * DMODEL + d] * cf[t * 8 + b] * g);
    }
    comb[i] = r;
    comb_bf[i] = (bf16)r;
}

// row softmax over 1024 keys; outputs bf16 probs.  grid=B*S, block=256
__global__ void softmax_rows_k(const float* __restrict__ scores, bf16* __restrict__ P)
{
    __shared__ float sm[256];
    const long row = blockIdx.x;
    const float* sp = scores + row * (long)SEQ;
    float v[4];
    float mx = -3.4e38f;
#pragma unroll
    for (int u = 0; u < 4; ++u) { v[u] = sp[threadIdx.x + u * 256]; mx = fmaxf(mx, v[u]); }
    sm[threadIdx.x] = mx;
    __syncthreads();
    for (int st = 128; st > 0; st >>= 1) {
        if (threadIdx.x < st) sm[threadIdx.x] = fmaxf(sm[threadIdx.x], sm[threadIdx.x + st]);
        __syncthreads();
    }
    mx = sm[0];
    __syncthreads();
    float s = 0.f;
#pragma unroll
    for (int u = 0; u < 4; ++u) { v[u] = __expf(v[u] - mx); s += v[u]; }
    sm[threadIdx.x] = s;
    __syncthreads();
    for (int st = 128; st > 0; st >>= 1) {
        if (threadIdx.x < st) sm[threadIdx.x] += sm[threadIdx.x + st];
        __syncthreads();
    }
    float invs = 1.f / sm[0];
    bf16* pp = P + row * (long)SEQ;
#pragma unroll
    for (int u = 0; u < 4; ++u) pp[threadIdx.x + u * 256] = (bf16)(v[u] * invs);
}

// comp = mem_gated * cfp[b]; layernorm(comp)*g+b -> bf16.  grid=B*S, block=256
__global__ void layernorm_k(const float* __restrict__ mg, const float* __restrict__ cfp,
                            const float* __restrict__ ln_g, const float* __restrict__ ln_b,
                            bf16* __restrict__ out)
{
    __shared__ float sm[256];
    const long row = blockIdx.x;
    const int b = (int)(row / SEQ);
    const float c = cfp[b];
    const float* xp = mg + row * (long)DMODEL;
    float v[4];
    float s = 0.f;
#pragma unroll
    for (int u = 0; u < 4; ++u) { v[u] = xp[threadIdx.x + u * 256] * c; s += v[u]; }
    sm[threadIdx.x] = s;
    __syncthreads();
    for (int st = 128; st > 0; st >>= 1) {
        if (threadIdx.x < st) sm[threadIdx.x] += sm[threadIdx.x + st];
        __syncthreads();
    }
    const float mu = sm[0] * (1.f / DMODEL);
    __syncthreads();
    float s2 = 0.f;
#pragma unroll
    for (int u = 0; u < 4; ++u) { float dd = v[u] - mu; s2 += dd * dd; }
    sm[threadIdx.x] = s2;
    __syncthreads();
    for (int st = 128; st > 0; st >>= 1) {
        if (threadIdx.x < st) sm[threadIdx.x] += sm[threadIdx.x + st];
        __syncthreads();
    }
    const float var = sm[0] * (1.f / DMODEL);
    const float inv = rsqrtf(var + 1e-5f);
    bf16* op = out + row * (long)DMODEL;
#pragma unroll
    for (int u = 0; u < 4; ++u) {
        int cix = threadIdx.x + u * 256;
        op[cix] = (bf16)((v[u] - mu) * inv * ln_g[cix] + ln_b[cix]);
    }
}

// ---------------------------------------------------------------------------
static inline int gemm_blocks(int M, int N, int NC)
{
    int strips = (M / 16) * ((N / 16) / NC);
    return (strips + 3) / 4;
}

extern "C" void kernel_launch(void* const* d_in, const int* in_sizes, int n_in,
                              void* d_out, int out_size, void* d_ws, size_t ws_size,
                              hipStream_t stream)
{
    (void)in_sizes; (void)n_in; (void)out_size; (void)ws_size;

    const float* x         = (const float*)d_in[0];
    const unsigned char* attn_mask = (const unsigned char*)d_in[1];
    const float* ts_logits = (const float*)d_in[2];
    const float* gw1       = (const float*)d_in[3];
    const float* gb1       = (const float*)d_in[4];
    const float* gw2       = (const float*)d_in[5];
    const float* gb2       = (const float*)d_in[6];
    const float* gweights  = (const float*)d_in[7];
    const float* ssm_A     = (const float*)d_in[8];
    const float* ssm_Bm    = (const float*)d_in[9];
    const float* ssm_C     = (const float*)d_in[10];
    const float* ssm_D     = (const float*)d_in[11];
    const float* cg_w1     = (const float*)d_in[12];
    const float* cg_b1     = (const float*)d_in[13];
    const float* cg_w2     = (const float*)d_in[14];
    const float* cg_b2     = (const float*)d_in[15];
    const float* wq = (const float*)d_in[16]; const float* bq = (const float*)d_in[17];
    const float* wk = (const float*)d_in[18]; const float* bk = (const float*)d_in[19];
    const float* wv = (const float*)d_in[20]; const float* bv = (const float*)d_in[21];
    const float* wo = (const float*)d_in[22]; const float* bo = (const float*)d_in[23];
    const float* mem_gate  = (const float*)d_in[24];
    const float* cp_w1     = (const float*)d_in[25];
    const float* cp_b1     = (const float*)d_in[26];
    const float* cp_w2     = (const float*)d_in[27];
    const float* cp_b2     = (const float*)d_in[28];
    const float* ln_g      = (const float*)d_in[29];
    const float* ln_b      = (const float*)d_in[30];
    const float* out_w     = (const float*)d_in[31];
    const float* out_b     = (const float*)d_in[32];
    float* out = (float*)d_out;

    // ---- workspace bump allocation with lifetime-based aliasing (~153 MB) ----
    char* ws = (char*)d_ws;
    size_t off = 0;
    auto take = [&](size_t bytes) -> char* {
        char* p = ws + off; off += (bytes + 255) & ~(size_t)255; return p;
    };
    const long EL = (long)BSROWS * DMODEL; // 8M elements
    bf16* x_bf     = (bf16*)take(EL * 2);            // later: K_bf
    bf16* w_gw1    = (bf16*)take(3L * DMODEL * DMODEL * 2);   // packed
    bf16* w_gw2    = (bf16*)take(3L * DMODEL * DMODEL * 2);   // packed
    bf16* w_q      = (bf16*)take((long)DMODEL * DMODEL * 2);  // packed
    bf16* w_k      = (bf16*)take((long)DMODEL * DMODEL * 2);  // packed
    bf16* w_v      = (bf16*)take((long)DMODEL * DMODEL * 2);  // packed
    bf16* w_o      = (bf16*)take((long)DMODEL * DMODEL * 2);  // packed
    bf16* w_ow     = (bf16*)take((long)DMODEL * DMODEL * 2);  // packed
    bf16* w_Bm     = (bf16*)take(3L * DMODEL * NST * 2);      // packed
    float* gatedF  = (float*)take(EL * 4);           // later: scores, mem_gated
    bf16* t1_bf    = (bf16*)take(EL * 2);            // later: Q_bf, attnT_bf
    bf16* gated_bf = (bf16*)take(EL * 2);            // later: V_bf(packed), normed_bf
    float* comb    = (float*)take(EL * 4);
    bf16* comb_bf  = (bf16*)take(EL * 2);            // later: P_bf
    float* U       = (float*)take(3L * BSROWS * NST * 4);
    float* Hs      = (float*)take(3L * BSROWS * NST * 4);
    float* meanG   = (float*)take((long)BATCHN * DMODEL * 4); // reused for mem mean
    float* cfv     = (float*)take(3L * BATCHN * 4);
    float* cfp     = (float*)take((long)BATCHN * 4);
    // aliases (lifetimes are strictly sequential)
    bf16* K_bf = x_bf;          float* scoresF  = gatedF;   float* memGF = gatedF;
    bf16* Q_bf = t1_bf;         bf16* attnT_bf  = t1_bf;
    bf16* V_bf = gated_bf;      bf16* normed_bf = gated_bf;
    bf16* P_bf = comb_bf;

    auto conv = [&](const float* src, bf16* dst, long n) {
        f32_to_bf16_k<<<(int)((n + 255) / 256), 256, 0, stream>>>(src, dst, n);
    };
    auto packB = [&](const float* src, bf16* dst, int K, int N, int slices) {
        const long n = (long)K * N;
        for (int t = 0; t < slices; ++t)
            packB_k<<<(int)((n + 255) / 256), 256, 0, stream>>>(src + t * n, dst + t * n, K, N);
    };

    // ---- 0. operand prep: x as row-major bf16 A; all weights packed ----
    conv(x, x_bf, EL);
    packB(gw1, w_gw1, DMODEL, DMODEL, 3);
    packB(gw2, w_gw2, DMODEL, DMODEL, 3);
    packB(wq, w_q, DMODEL, DMODEL, 1);
    packB(wk, w_k, DMODEL, DMODEL, 1);
    packB(wv, w_v, DMODEL, DMODEL, 1);
    packB(wo, w_o, DMODEL, DMODEL, 1);
    packB(out_w, w_ow, DMODEL, DMODEL, 1);
    packB(ssm_Bm, w_Bm, DMODEL, NST, 3);

    // ---- 1. multi-gate mixing: gated = sum_i x * sigmoid(silu(x@gw1+b)@gw2+b)*gwt[i]
    for (int i = 0; i < 3; ++i) {
        gemm_wmma<EP_SILU_BF16, 0, false, 4><<<gemm_blocks(BSROWS, DMODEL, 4), 128, 0, stream>>>(
            x_bf, w_gw1 + (long)i * DMODEL * DMODEL, nullptr, t1_bf, gb1 + i * DMODEL,
            nullptr, nullptr, nullptr,
            BSROWS, DMODEL, DMODEL, DMODEL, 0, DMODEL, 0, 0, 0, 0, 1.f, 0);
        gemm_wmma<EP_GATE_ACC, 0, false, 4><<<gemm_blocks(BSROWS, DMODEL, 4), 128, 0, stream>>>(
            t1_bf, w_gw2 + (long)i * DMODEL * DMODEL, gatedF, nullptr, gb2 + i * DMODEL,
            x, gweights + i, nullptr,
            BSROWS, DMODEL, DMODEL, DMODEL, 0, DMODEL, 0, 0, 0, 0, 1.f, i > 0);
    }

    // ---- 2. SSM conditioning gates ----
    conv(gatedF, gated_bf, EL);
    mean_seq_k<<<(BATCHN * DMODEL + 255) / 256, 256, 0, stream>>>(gatedF, meanG);
    gate_mlp_k<<<dim3(BATCHN, 3), 256, 256 * sizeof(float), stream>>>(
        meanG, cg_w1, cg_b1, cg_w2, cg_b2, cfv, 256);

    // ---- 3. U[t] = gated @ Bm[t]  (cf folded later as a scalar) ----
    for (int t = 0; t < 3; ++t)
        gemm_wmma<EP_F32, 0, false, 1><<<gemm_blocks(BSROWS, NST, 1), 128, 0, stream>>>(
            gated_bf, w_Bm + (long)t * DMODEL * NST, U + (long)t * BSROWS * NST, nullptr,
            nullptr, nullptr, nullptr, nullptr,
            BSROWS, NST, DMODEL, DMODEL, 0, NST, 0, 0, 0, 0, 1.f, 0);

    // ---- 4. sequential scan + timescale combine ----
    ssm_scan_k<<<24, 32, 0, stream>>>(U, ssm_A, cfv, Hs);
    ssm_combine_k<<<(int)((EL + 255) / 256), 256, 0, stream>>>(
        Hs, ssm_C, ssm_D, cfv, gatedF, ts_logits, comb, comb_bf);

    // ---- 5. Q,K (row-major bf16) and V (packed-B layout) projections ----
    gemm_wmma<EP_BF16, 0, false, 4><<<gemm_blocks(BSROWS, DMODEL, 4), 128, 0, stream>>>(
        comb_bf, w_q, nullptr, Q_bf, bq, nullptr, nullptr, nullptr,
        BSROWS, DMODEL, DMODEL, DMODEL, 0, DMODEL, 0, 0, 0, 0, 1.f, 0);
    gemm_wmma<EP_BF16, 0, false, 4><<<gemm_blocks(BSROWS, DMODEL, 4), 128, 0, stream>>>(
        comb_bf, w_k, nullptr, K_bf, bk, nullptr, nullptr, nullptr,
        BSROWS, DMODEL, DMODEL, DMODEL, 0, DMODEL, 0, 0, 0, 0, 1.f, 0);
    gemm_wmma<EP_PACKV, 0, false, 4><<<gemm_blocks(BSROWS, DMODEL, 4), 128, 0, stream>>>(
        comb_bf, w_v, nullptr, V_bf, bv, nullptr, nullptr, nullptr,
        BSROWS, DMODEL, DMODEL, DMODEL, 0, DMODEL, 0, 0, 0, 0, 1.f, 0);

    // ---- 6. scores = mask ? (Q@K^T)/sqrt(D) : -1e30  (batched NT GEMM) ----
    gemm_wmma<EP_MASK, 1, false, 4><<<dim3(gemm_blocks(SEQ, SEQ, 4), 1, BATCHN), 128, 0, stream>>>(
        Q_bf, K_bf, scoresF, nullptr, nullptr, nullptr, nullptr, attn_mask,
        SEQ, SEQ, DMODEL, DMODEL, DMODEL, SEQ,
        (long)SEQ * DMODEL, (long)SEQ * DMODEL, (long)SEQ * SEQ, (long)SEQ * SEQ,
        0.03125f /* 1/sqrt(1024) */, 0);

    // ---- 7. softmax -> bf16 probs ----
    softmax_rows_k<<<BATCHN * SEQ, 256, 0, stream>>>(scoresF, P_bf);

    // ---- 8. attn = P @ V(packed), stored transposed (fuses swapaxes shuffle) ----
    gemm_wmma<EP_BF16, 0, true, 4><<<dim3(gemm_blocks(SEQ, DMODEL, 4), 1, BATCHN), 128, 0, stream>>>(
        P_bf, V_bf, nullptr, attnT_bf, nullptr, nullptr, nullptr, nullptr,
        SEQ, DMODEL, SEQ, SEQ, 0, SEQ /* ldc of transposed store */,
        (long)SEQ * SEQ, (long)SEQ * DMODEL, (long)SEQ * DMODEL, 0, 1.f, 0);

    // ---- 9. mem_gated = (attnT @ wo + bo) * comb * mem_gate ----
    gemm_wmma<EP_MEMGATE, 0, false, 4><<<gemm_blocks(BSROWS, DMODEL, 4), 128, 0, stream>>>(
        attnT_bf, w_o, memGF, nullptr, bo, comb, mem_gate, nullptr,
        BSROWS, DMODEL, DMODEL, DMODEL, 0, DMODEL, 0, 0, 0, 0, 1.f, 0);

    // ---- 10. compression gate + layernorm ----
    mean_seq_k<<<(BATCHN * DMODEL + 255) / 256, 256, 0, stream>>>(memGF, meanG);
    gate_mlp_k<<<dim3(BATCHN, 1), 512, 512 * sizeof(float), stream>>>(
        meanG, cp_w1, cp_b1, cp_w2, cp_b2, cfp, 512);
    layernorm_k<<<BATCHN * SEQ, 256, 0, stream>>>(memGF, cfp, ln_g, ln_b, normed_bf);

    // ---- 11. output projection ----
    gemm_wmma<EP_F32, 0, false, 4><<<gemm_blocks(BSROWS, DMODEL, 4), 128, 0, stream>>>(
        normed_bf, w_ow, out, nullptr, out_b, nullptr, nullptr, nullptr,
        BSROWS, DMODEL, DMODEL, DMODEL, 0, DMODEL, 0, 0, 0, 0, 1.f, 0);
}
